// ISONET_87694642250044
// MI455X (gfx1250) — matrix operand order repaired and checked by reading, not compiled, not run on hardware
//
#include <hip/hip_runtime.h>

// ---------------------------------------------------------------------------
// Problem constants (from reference)
// ---------------------------------------------------------------------------
#define NG   1024        // NUM_GRAPHS
#define NPG  64          // NODES_PER_GRAPH
#define ME   256         // MAX_EDGES
#define DD   128         // D
#define MH   256         // MSG_H
#define NODE_IN 32
#define EDGE_IN 16
#define PROP_STEPS 5
#define SINK_ITERS 20
#define INV_TEMP 100.0f  // 1/SINK_TEMP

typedef __attribute__((ext_vector_type(16))) __bf16 v16bf;
typedef __attribute__((ext_vector_type(8)))  float  v8f;

// ---------------------------------------------------------------------------
// WMMA helpers (CDNA5 wave32, V_WMMA_F32_16X16X32_BF16)
// ---------------------------------------------------------------------------
__device__ __forceinline__ v8f wmma_bf16(v16bf a, v16bf b, v8f c) {
    return __builtin_amdgcn_wmma_f32_16x16x32_bf16(
        /*neg_a=*/false, a, /*neg_b=*/false, b,
        /*c_mod=*/(short)0, c, /*reuse_a=*/false, /*reuse_b=*/false);
}

__device__ __forceinline__ v8f splat8(float x) {
    v8f v;
#pragma unroll
    for (int i = 0; i < 8; ++i) v[i] = x;
    return v;
}

// A tile (16x32, bf16) from a row-major matrix (rows = M).
// Contiguous K order (same as B loaders): lanes 0-15 -> M=lane, halfs0..15 K=kb..kb+15
//             lanes16-31 -> M=lane-16, K=kb+16..kb+31.
__device__ __forceinline__ v16bf load_a_rows(const __bf16* mat, int ld, int row0, int kb) {
    int lane = threadIdx.x & 31;
    const __bf16* p = mat + (size_t)(row0 + (lane & 15)) * ld + kb + ((lane & 16) ? 16 : 0);
    v16bf a;
#pragma unroll
    for (int i = 0; i < 8; ++i) a[i] = p[i];
#pragma unroll
    for (int i = 0; i < 8; ++i) a[8 + i] = p[8 + i];
    return a;
}

// B tile (32x16) where B = Y^T and we have Y rows (row-major, ld elems):
// lane holds column n (= Y row nb+lane&15), 16 consecutive k halves.
__device__ __forceinline__ v16bf load_bt_rows(const __bf16* mat, int ld, int nb, int kb) {
    int lane = threadIdx.x & 31;
    const __bf16* p = mat + (size_t)(nb + (lane & 15)) * ld + kb + ((lane & 16) ? 16 : 0);
    v16bf b;
#pragma unroll
    for (int i = 0; i < 16; ++i) b[i] = p[i];
    return b;
}

// B tile from pre-shuffled weight buffer: one contiguous 32B load per lane.
__device__ __forceinline__ v16bf load_b_tiled(const __bf16* wt, int ntiles, int kt, int nt) {
    const __bf16* p = wt + ((((size_t)kt * ntiles) + nt) * 32 + (threadIdx.x & 31)) * 16;
    return *(const v16bf*)p;
}

// B tile gathered column-major from a row-major matrix (B[k][n] = mat[k][n]).
__device__ __forceinline__ v16bf load_b_colmaj(const __bf16* mat, int ld, int kb, int nb) {
    int lane = threadIdx.x & 31;
    int n  = nb + (lane & 15);
    int k0 = kb + ((lane & 16) ? 16 : 0);
    v16bf b;
#pragma unroll
    for (int i = 0; i < 16; ++i) b[i] = mat[(size_t)(k0 + i) * ld + n];
    return b;
}

// A tile from f32 LDS with exp() applied on the fly (plan = exp(log_alpha)).
__device__ __forceinline__ v16bf load_a_exp(const float* la, int ld, int row0, int kb) {
    int lane = threadIdx.x & 31;
    const float* p = la + (size_t)(row0 + (lane & 15)) * ld + kb + ((lane & 16) ? 16 : 0);
    v16bf a;
#pragma unroll
    for (int i = 0; i < 8; ++i) a[i] = (__bf16)__expf(p[i]);
#pragma unroll
    for (int i = 0; i < 8; ++i) a[8 + i] = (__bf16)__expf(p[8 + i]);
    return a;
}

// ---------------------------------------------------------------------------
// Weight pre-shuffle: row-major f32 (K x N) -> tiled bf16 B-operand layout
// ---------------------------------------------------------------------------
__global__ void k_prep_w(const float* __restrict__ W, __bf16* __restrict__ Wt, int K, int N) {
    int idx = blockIdx.x * blockDim.x + threadIdx.x;
    if (idx >= K * N) return;
    int k = idx / N, n = idx - k * N;
    int kt = k >> 5, kk = k & 31;
    int nt = n >> 4, nn = n & 15;
    int lanev = (kk < 16) ? nn : (nn + 16);
    int h = kk & 15;
    int ntiles = N >> 4;
    Wt[((((size_t)kt * ntiles) + nt) * 32 + lanev) * 16 + h] = (__bf16)W[idx];
}

// ---------------------------------------------------------------------------
// Encoders (tiny K, plain VALU is fine)
// ---------------------------------------------------------------------------
__global__ void k_encode_nodes(const float* __restrict__ nf, const float* __restrict__ Wn,
                               const float* __restrict__ bn, float* __restrict__ node,
                               __bf16* __restrict__ node_bf, int Nn) {
    int idx = blockIdx.x * blockDim.x + threadIdx.x;
    if (idx >= Nn * DD) return;
    int i = idx / DD, d = idx - i * DD;
    float acc = bn[d];
#pragma unroll
    for (int k = 0; k < NODE_IN; ++k) acc += nf[i * NODE_IN + k] * Wn[k * DD + d];
    node[idx] = acc;
    node_bf[idx] = (__bf16)acc;
}

__global__ void k_encode_edges(const float* __restrict__ ef, const float* __restrict__ We,
                               const float* __restrict__ be, __bf16* __restrict__ edge_bf, int E) {
    int idx = blockIdx.x * blockDim.x + threadIdx.x;
    if (idx >= E * DD) return;
    int e = idx / DD, d = idx - e * DD;
    float acc = be[d];
#pragma unroll
    for (int k = 0; k < EDGE_IN; ++k) acc += ef[e * EDGE_IN + k] * We[k * DD + d];
    edge_bf[idx] = (__bf16)acc;
}

__global__ void k_cvt_f32_bf16(const float* __restrict__ src, __bf16* __restrict__ dst, long long n) {
    long long idx = (long long)blockIdx.x * blockDim.x + threadIdx.x;
    if (idx < n) dst[idx] = (__bf16)src[idx];
}

// ---------------------------------------------------------------------------
// Message kernel: per block = 16 edges, 8 waves.
// X = [src|dst|edge] (16x384 bf16 in LDS). Layer1 (K=384,N=256) fwd uses Wm1,
// bwd uses Wr1 with K blocks remapped (dst,src,edge). Layer2 (K=256,N=128).
// mode 0: atomic segment-sum into agg; mode 1: atomic add (m_f+m_b) into
// stacked[edge_slot].
// ---------------------------------------------------------------------------
__global__ __launch_bounds__(256) void k_message(
    const __bf16* __restrict__ node_bf, const __bf16* __restrict__ edge_bf,
    const __bf16* __restrict__ Wm1t, const float* __restrict__ bm1,
    const __bf16* __restrict__ Wm2t, const float* __restrict__ bm2,
    const __bf16* __restrict__ Wr1t, const float* __restrict__ br1,
    const __bf16* __restrict__ Wr2t, const float* __restrict__ br2,
    const int* __restrict__ from_idx, const int* __restrict__ to_idx,
    const int* __restrict__ edge_slot,
    float* __restrict__ agg, float* __restrict__ stacked, int E, int mode)
{
    __shared__ __bf16 xs[16][3 * DD + 8];
    __shared__ __bf16 hs[2][16][MH + 8];

    int wave = threadIdx.x >> 5, lane = threadIdx.x & 31;
    int e0 = blockIdx.x * 16;

    // cooperative gather of src/dst/edge rows (128 bf16 = 64 dwords each)
    for (int r = wave; r < 16; r += 8) {
        int e = e0 + r;
        if (e >= E) e = E - 1;
        const unsigned* ps = (const unsigned*)(node_bf + (size_t)from_idx[e] * DD);
        const unsigned* pd = (const unsigned*)(node_bf + (size_t)to_idx[e] * DD);
        const unsigned* pe = (const unsigned*)(edge_bf + (size_t)e * DD);
        unsigned* xr = (unsigned*)&xs[r][0];
        xr[lane]       = ps[lane];       xr[lane + 32]       = ps[lane + 32];
        xr[64 + lane]  = pd[lane];       xr[64 + lane + 32]  = pd[lane + 32];
        xr[128 + lane] = pe[lane];       xr[128 + lane + 32] = pe[lane + 32];
    }
    __syncthreads();

    // Layer 1: 32 (dir x ntile) tiles over 8 waves
#pragma unroll
    for (int i = 0; i < 4; ++i) {
        int t = wave * 4 + i;
        int dir = t >> 4, nt = t & 15;
        const __bf16* W1 = dir ? Wr1t : Wm1t;
        const float*  B1 = dir ? br1  : bm1;
        v8f acc = splat8(B1[nt * 16 + (lane & 15)]);
#pragma unroll
        for (int kt = 0; kt < 12; ++kt) {
            int kb = kt * 32;
            int kbx = dir ? (kb < 128 ? kb + 128 : (kb < 256 ? kb - 128 : kb)) : kb;
            v16bf a = load_a_rows(&xs[0][0], 3 * DD + 8, 0, kbx);
            v16bf b = load_b_tiled(W1, 16, kt, nt);
            acc = wmma_bf16(a, b, acc);
        }
        int m0 = (lane & 16) ? 8 : 0, n = lane & 15;
#pragma unroll
        for (int r = 0; r < 8; ++r) {
            float v = acc[r];
            hs[dir][m0 + r][nt * 16 + n] = (__bf16)(v > 0.f ? v : 0.f);
        }
    }
    __syncthreads();

    // Layer 2: 16 (dir x ntile) tiles over 8 waves
#pragma unroll
    for (int i = 0; i < 2; ++i) {
        int t = wave * 2 + i;
        int dir = t >> 3, nt = t & 7;
        const __bf16* W2 = dir ? Wr2t : Wm2t;
        const float*  B2 = dir ? br2  : bm2;
        v8f acc = splat8(B2[nt * 16 + (lane & 15)]);
#pragma unroll
        for (int kt = 0; kt < 8; ++kt) {
            v16bf a = load_a_rows(&hs[dir][0][0], MH + 8, 0, kt * 32);
            v16bf b = load_b_tiled(W2, 8, kt, nt);
            acc = wmma_bf16(a, b, acc);
        }
        int m0 = (lane & 16) ? 8 : 0, n = nt * 16 + (lane & 15);
#pragma unroll
        for (int r = 0; r < 8; ++r) {
            int e = e0 + m0 + r;
            if (e < E) {
                if (mode == 0) {
                    int tgt = dir ? from_idx[e] : to_idx[e];
                    atomicAdd(&agg[(size_t)tgt * DD + n], acc[r]);
                } else {
                    atomicAdd(&stacked[(size_t)edge_slot[e] * DD + n], acc[r]);
                }
            }
        }
    }
}

// ---------------------------------------------------------------------------
// Node update: per block = 16 nodes. X = [node|agg] (16x256), K=256 -> N=256
// relu -> K=256 -> N=128, residual add, refresh bf16 copy.
// ---------------------------------------------------------------------------
__global__ __launch_bounds__(256) void k_update(
    float* __restrict__ node, __bf16* __restrict__ node_bf, const float* __restrict__ agg,
    const __bf16* __restrict__ Wu1t, const float* __restrict__ bu1,
    const __bf16* __restrict__ Wu2t, const float* __restrict__ bu2)
{
    __shared__ __bf16 xs[16][2 * DD + 8];
    __shared__ __bf16 hs[16][MH + 8];

    int wave = threadIdx.x >> 5, lane = threadIdx.x & 31;
    int i0 = blockIdx.x * 16;

    for (int r = wave; r < 16; r += 8) {
        int i = i0 + r;
        const unsigned* pn = (const unsigned*)(node_bf + (size_t)i * DD);
        unsigned* xr = (unsigned*)&xs[r][0];
        xr[lane] = pn[lane];
        xr[lane + 32] = pn[lane + 32];
        const float* pa = agg + (size_t)i * DD;
#pragma unroll
        for (int jj = 0; jj < 4; ++jj)
            xs[r][DD + lane + 32 * jj] = (__bf16)pa[lane + 32 * jj];
    }
    __syncthreads();

#pragma unroll
    for (int i = 0; i < 2; ++i) {
        int nt = wave * 2 + i;
        v8f acc = splat8(bu1[nt * 16 + (lane & 15)]);
#pragma unroll
        for (int kt = 0; kt < 8; ++kt) {
            v16bf a = load_a_rows(&xs[0][0], 2 * DD + 8, 0, kt * 32);
            v16bf b = load_b_tiled(Wu1t, 16, kt, nt);
            acc = wmma_bf16(a, b, acc);
        }
        int m0 = (lane & 16) ? 8 : 0, n = lane & 15;
#pragma unroll
        for (int r = 0; r < 8; ++r) {
            float v = acc[r];
            hs[m0 + r][nt * 16 + n] = (__bf16)(v > 0.f ? v : 0.f);
        }
    }
    __syncthreads();

    {
        int nt = wave;  // 8 waves, 8 n-tiles of 128
        v8f acc = splat8(bu2[nt * 16 + (lane & 15)]);
#pragma unroll
        for (int kt = 0; kt < 8; ++kt) {
            v16bf a = load_a_rows(&hs[0][0], MH + 8, 0, kt * 32);
            v16bf b = load_b_tiled(Wu2t, 8, kt, nt);
            acc = wmma_bf16(a, b, acc);
        }
        int m0 = (lane & 16) ? 8 : 0, n = nt * 16 + (lane & 15);
#pragma unroll
        for (int r = 0; r < 8; ++r) {
            size_t idx = (size_t)(i0 + m0 + r) * DD + n;
            float v = node[idx] + acc[r];
            node[idx] = v;
            node_bf[idx] = (__bf16)v;
        }
    }
}

// ---------------------------------------------------------------------------
// t = relu(stacked @ Ws1 + bs1) @ Ws2 + bs2, masked per-graph row count.
// per block = 16 rows of the (NG*ME) x 128 input.
// ---------------------------------------------------------------------------
__global__ __launch_bounds__(256) void k_tmlp(
    const __bf16* __restrict__ xin,
    const __bf16* __restrict__ Ws1t, const float* __restrict__ bs1,
    const __bf16* __restrict__ Ws2t, const float* __restrict__ bs2,
    const int* __restrict__ counts, __bf16* __restrict__ tout)
{
    __shared__ __bf16 hs[16][MH + 8];
    int wave = threadIdx.x >> 5, lane = threadIdx.x & 31;
    int r0 = blockIdx.x * 16;

#pragma unroll
    for (int i = 0; i < 2; ++i) {
        int nt = wave * 2 + i;
        v8f acc = splat8(bs1[nt * 16 + (lane & 15)]);
#pragma unroll
        for (int kt = 0; kt < 4; ++kt) {
            v16bf a = load_a_rows(xin, DD, r0, kt * 32);
            v16bf b = load_b_tiled(Ws1t, 16, kt, nt);
            acc = wmma_bf16(a, b, acc);
        }
        int m0 = (lane & 16) ? 8 : 0, n = lane & 15;
#pragma unroll
        for (int r = 0; r < 8; ++r) {
            float v = acc[r];
            hs[m0 + r][nt * 16 + n] = (__bf16)(v > 0.f ? v : 0.f);
        }
    }
    __syncthreads();

#pragma unroll
    for (int i = 0; i < 2; ++i) {
        int nt = wave * 2 + i;
        v8f acc = splat8(bs2[nt * 16 + (lane & 15)]);
#pragma unroll
        for (int kt = 0; kt < 8; ++kt) {
            v16bf a = load_a_rows(&hs[0][0], MH + 8, 0, kt * 32);
            v16bf b = load_b_tiled(Ws2t, 16, kt, nt);
            acc = wmma_bf16(a, b, acc);
        }
        int m0 = (lane & 16) ? 8 : 0, n = nt * 16 + (lane & 15);
#pragma unroll
        for (int r = 0; r < 8; ++r) {
            int gr = r0 + m0 + r;
            int g = gr >> 8, s = gr & 255;
            float v = (s < counts[g]) ? acc[r] : 0.f;
            tout[(size_t)gr * ME + n] = (__bf16)v;
        }
    }
}

// ---------------------------------------------------------------------------
// Fused Sinkhorn + transport per graph-pair. One block per pair, 16 waves.
// log_alpha (256x256 f32, row stride 257 for bank-conflict-free columns)
// lives entirely in LDS (~263 KB of the WGP's 320 KB).
// ---------------------------------------------------------------------------
#define LA_LD 257

__global__ __launch_bounds__(512) void k_sinkhorn(
    const __bf16* __restrict__ t_bf,
    const __bf16* __restrict__ stacked_bf,
    const float* __restrict__ stacked_f,
    float* __restrict__ out)
{
    extern __shared__ float la[];      // [256][LA_LD]
    __shared__ float red[16];

    int wave = threadIdx.x >> 5, lane = threadIdx.x & 31;
    int b = blockIdx.x;
    const __bf16* tq = t_bf + (size_t)(2 * b) * ME * ME;
    const __bf16* tc = t_bf + (size_t)(2 * b + 1) * ME * ME;

    // 1. log_alpha = (tq @ tc^T) / temp  (256 WMMA tiles over 16 waves)
    for (int t = wave; t < 256; t += 16) {
        int ti = t >> 4, tj = t & 15;
        v8f acc = splat8(0.f);
#pragma unroll
        for (int kt = 0; kt < 8; ++kt) {
            v16bf a  = load_a_rows(tq, ME, ti * 16, kt * 32);
            v16bf bb = load_bt_rows(tc, ME, tj * 16, kt * 32);
            acc = wmma_bf16(a, bb, acc);
        }
        int m0 = (lane & 16) ? 8 : 0, n = tj * 16 + (lane & 15);
#pragma unroll
        for (int r = 0; r < 8; ++r)
            la[(size_t)(ti * 16 + m0 + r) * LA_LD + n] = acc[r] * INV_TEMP;
    }
    __syncthreads();

    // 2. Sinkhorn iterations entirely in LDS
    for (int it = 0; it < SINK_ITERS; ++it) {
        // row normalization (axis=2)
        for (int i = wave; i < 256; i += 16) {
            float* row = la + (size_t)i * LA_LD;
            float v[8], m = -1e30f;
#pragma unroll
            for (int jj = 0; jj < 8; ++jj) { v[jj] = row[lane + 32 * jj]; m = fmaxf(m, v[jj]); }
            for (int o = 16; o > 0; o >>= 1) m = fmaxf(m, __shfl_xor(m, o, 32));
            float s = 0.f;
#pragma unroll
            for (int jj = 0; jj < 8; ++jj) s += __expf(v[jj] - m);
            for (int o = 16; o > 0; o >>= 1) s += __shfl_xor(s, o, 32);
            float lse = m + __logf(s);
#pragma unroll
            for (int jj = 0; jj < 8; ++jj) row[lane + 32 * jj] = v[jj] - lse;
        }
        __syncthreads();
        // column normalization (axis=1)
        for (int j = wave; j < 256; j += 16) {
            float v[8], m = -1e30f;
#pragma unroll
            for (int ii = 0; ii < 8; ++ii) { v[ii] = la[(size_t)(lane + 32 * ii) * LA_LD + j]; m = fmaxf(m, v[ii]); }
            for (int o = 16; o > 0; o >>= 1) m = fmaxf(m, __shfl_xor(m, o, 32));
            float s = 0.f;
#pragma unroll
            for (int ii = 0; ii < 8; ++ii) s += __expf(v[ii] - m);
            for (int o = 16; o > 0; o >>= 1) s += __shfl_xor(s, o, 32);
            float lse = m + __logf(s);
#pragma unroll
            for (int ii = 0; ii < 8; ++ii) la[(size_t)(lane + 32 * ii) * LA_LD + j] = v[ii] - lse;
        }
        __syncthreads();
    }

    // 3. out = sum relu(q - exp(la) @ c); plan@c via WMMA (exp in A loader)
    const float*  qf = stacked_f  + (size_t)(2 * b)     * ME * DD;
    const __bf16* cb = stacked_bf + (size_t)(2 * b + 1) * ME * DD;
    float partial = 0.f;
    for (int t = wave; t < 128; t += 16) {  // 16 i-tiles x 8 d-tiles
        int ti = t >> 3, td = t & 7;
        v8f acc = splat8(0.f);
#pragma unroll
        for (int kt = 0; kt < 8; ++kt) {
            v16bf a  = load_a_exp(la, LA_LD, ti * 16, kt * 32);
            v16bf bb = load_b_colmaj(cb, DD, kt * 32, td * 16);
            acc = wmma_bf16(a, bb, acc);
        }
        int m0 = (lane & 16) ? 8 : 0, n = td * 16 + (lane & 15);
#pragma unroll
        for (int r = 0; r < 8; ++r) {
            float q = qf[(size_t)(ti * 16 + m0 + r) * DD + n];
            float v = q - acc[r];
            partial += v > 0.f ? v : 0.f;
        }
    }
    for (int o = 16; o > 0; o >>= 1) partial += __shfl_xor(partial, o, 32);
    if (lane == 0) red[wave] = partial;
    __syncthreads();
    if (threadIdx.x == 0) {
        float s = 0.f;
        for (int w = 0; w < 16; ++w) s += red[w];
        out[b] = s;
    }
}

// ---------------------------------------------------------------------------
// Host launcher
// ---------------------------------------------------------------------------
extern "C" void kernel_launch(void* const* d_in, const int* in_sizes, int n_in,
                              void* d_out, int out_size, void* d_ws, size_t ws_size,
                              hipStream_t stream) {
    const float* node_features = (const float*)d_in[0];
    const float* edge_features = (const float*)d_in[1];
    const float* Wn  = (const float*)d_in[2];
    const float* bn  = (const float*)d_in[3];
    const float* We  = (const float*)d_in[4];
    const float* be  = (const float*)d_in[5];
    const float* Wm1 = (const float*)d_in[6];
    const float* bm1 = (const float*)d_in[7];
    const float* Wm2 = (const float*)d_in[8];
    const float* bm2 = (const float*)d_in[9];
    const float* Wr1 = (const float*)d_in[10];
    const float* br1 = (const float*)d_in[11];
    const float* Wr2 = (const float*)d_in[12];
    const float* br2 = (const float*)d_in[13];
    const float* Wu1 = (const float*)d_in[14];
    const float* bu1 = (const float*)d_in[15];
    const float* Wu2 = (const float*)d_in[16];
    const float* bu2 = (const float*)d_in[17];
    const float* Ws1 = (const float*)d_in[18];
    const float* bs1 = (const float*)d_in[19];
    const float* Ws2 = (const float*)d_in[20];
    const float* bs2 = (const float*)d_in[21];
    const int* from_idx   = (const int*)d_in[22];
    const int* to_idx     = (const int*)d_in[23];
    const int* edge_slot  = (const int*)d_in[24];
    const int* edge_counts= (const int*)d_in[25];

    const int Nn = in_sizes[0] / NODE_IN;   // 65536
    const int E  = in_sizes[1] / EDGE_IN;

    // ---- workspace bump allocator ----
    char* ws = (char*)d_ws;
    size_t off = 0;
    auto alloc = [&](size_t bytes) {
        void* p = ws + off;
        off += bytes;
        off = (off + 255) & ~(size_t)255;
        return p;
    };
    float*  node       = (float*) alloc((size_t)Nn * DD * 4);
    __bf16* node_bf    = (__bf16*)alloc((size_t)Nn * DD * 2);
    float*  agg        = (float*) alloc((size_t)Nn * DD * 4);
    __bf16* edge_bf    = (__bf16*)alloc((size_t)E * DD * 2);
    float*  stacked    = (float*) alloc((size_t)NG * ME * DD * 4);
    __bf16* stacked_bf = (__bf16*)alloc((size_t)NG * ME * DD * 2);
    __bf16* t_bf       = (__bf16*)alloc((size_t)NG * ME * ME * 2);
    __bf16* Wm1t = (__bf16*)alloc((size_t)3 * DD * MH * 2);
    __bf16* Wm2t = (__bf16*)alloc((size_t)MH * DD * 2);
    __bf16* Wr1t = (__bf16*)alloc((size_t)3 * DD * MH * 2);
    __bf16* Wr2t = (__bf16*)alloc((size_t)MH * DD * 2);
    __bf16* Wu1t = (__bf16*)alloc((size_t)2 * DD * MH * 2);
    __bf16* Wu2t = (__bf16*)alloc((size_t)MH * DD * 2);
    __bf16* Ws1t = (__bf16*)alloc((size_t)DD * ME * 2);
    __bf16* Ws2t = (__bf16*)alloc((size_t)ME * ME * 2);
    (void)ws_size; (void)n_in; (void)out_size;

    auto g1 = [](long long n, int b) { return (unsigned)((n + b - 1) / b); };

    // ---- weight shuffling ----
    k_prep_w<<<g1(3 * DD * MH, 256), 256, 0, stream>>>(Wm1, Wm1t, 3 * DD, MH);
    k_prep_w<<<g1(MH * DD, 256),     256, 0, stream>>>(Wm2, Wm2t, MH, DD);
    k_prep_w<<<g1(3 * DD * MH, 256), 256, 0, stream>>>(Wr1, Wr1t, 3 * DD, MH);
    k_prep_w<<<g1(MH * DD, 256),     256, 0, stream>>>(Wr2, Wr2t, MH, DD);
    k_prep_w<<<g1(2 * DD * MH, 256), 256, 0, stream>>>(Wu1, Wu1t, 2 * DD, MH);
    k_prep_w<<<g1(MH * DD, 256),     256, 0, stream>>>(Wu2, Wu2t, MH, DD);
    k_prep_w<<<g1(DD * ME, 256),     256, 0, stream>>>(Ws1, Ws1t, DD, ME);
    k_prep_w<<<g1(ME * ME, 256),     256, 0, stream>>>(Ws2, Ws2t, ME, ME);

    // ---- encoders ----
    k_encode_nodes<<<g1((long long)Nn * DD, 256), 256, 0, stream>>>(
        node_features, Wn, bn, node, node_bf, Nn);
    k_encode_edges<<<g1((long long)E * DD, 256), 256, 0, stream>>>(
        edge_features, We, be, edge_bf, E);

    // ---- propagation steps ----
    unsigned eblocks = g1(E, 16);
    for (int step = 0; step < PROP_STEPS; ++step) {
        hipMemsetAsync(agg, 0, (size_t)Nn * DD * 4, stream);
        k_message<<<eblocks, 256, 0, stream>>>(
            node_bf, edge_bf, Wm1t, bm1, Wm2t, bm2, Wr1t, br1, Wr2t, br2,
            from_idx, to_idx, edge_slot, agg, stacked, E, /*mode=*/0);
        k_update<<<Nn / 16, 256, 0, stream>>>(node, node_bf, agg, Wu1t, bu1, Wu2t, bu2);
    }

    // ---- final edge encoding, scattered into stacked ----
    hipMemsetAsync(stacked, 0, (size_t)NG * ME * DD * 4, stream);
    k_message<<<eblocks, 256, 0, stream>>>(
        node_bf, edge_bf, Wm1t, bm1, Wm2t, bm2, Wr1t, br1, Wr2t, br2,
        from_idx, to_idx, edge_slot, agg, stacked, E, /*mode=*/1);

    k_cvt_f32_bf16<<<g1((long long)NG * ME * DD, 256), 256, 0, stream>>>(
        stacked, stacked_bf, (long long)NG * ME * DD);

    // ---- Ws MLP with per-graph masking ----
    k_tmlp<<<(NG * ME) / 16, 256, 0, stream>>>(
        stacked_bf, Ws1t, bs1, Ws2t, bs2, edge_counts, t_bf);

    // ---- fused Sinkhorn + transport, one block per pair ----
    k_sinkhorn<<<NG / 2, 512, (size_t)ME * LA_LD * sizeof(float), stream>>>(
        t_bf, stacked_bf, stacked, (float*)d_out);
}
